// GCNStacker_89730456748329
// MI455X (gfx1250) — matrix operand order repaired and checked by reading, not compiled, not run on hardware
//
#include <hip/hip_runtime.h>
#include <hip/hip_bf16.h>

// ---------------------------------------------------------------------------
// CDNA5 (gfx1250) implementation of the 3-direction residual GCN stacker.
// Matrix work goes through v_wmma_f32_16x16x32_bf16 (wave32, 16x16 tiles).
// ---------------------------------------------------------------------------

typedef __attribute__((ext_vector_type(16))) __bf16 v16bf;
typedef __attribute__((ext_vector_type(8)))  float  v8f;

// Native f32 -> bf16 conversion: lowers to gfx1250 packed cvt ops (RNE),
// matching WMMA's internal rounding. Avoids the integer add/shift path.
__device__ __forceinline__ __bf16 f2bf(float f) { return (__bf16)f; }

// ---------------------------------------------------------------------------
__global__ void k_fill(float* __restrict__ p, float v, long n) {
  long i = blockIdx.x * (long)blockDim.x + threadIdx.x;
  if (i < n) p[i] = v;
}

// agg buffer layout: [b][node][64]; [0:32]=sum (init 0), [32:64]=max (init -inf)
__global__ void k_agg_init(float* __restrict__ agg, long n) {
  long i = blockIdx.x * (long)blockDim.x + threadIdx.x;
  if (i < n) agg[i] = ((i & 63) < 32) ? 0.0f : -__builtin_inff();
}

__global__ void k_count(const int* __restrict__ dst, int E, float* __restrict__ cnt) {
  int e = blockIdx.x * blockDim.x + threadIdx.x;
  if (e < E) atomicAdd(&cnt[dst[e]], 1.0f);
}

// scatter: one thread per (edge, batch); 32 features add + max
__global__ void k_scatter(const float* __restrict__ x,
                          const int* __restrict__ dst, const int* __restrict__ src,
                          int E, int NOUT, long total, float* __restrict__ agg) {
  long t = blockIdx.x * (long)blockDim.x + threadIdx.x;
  if (t >= total) return;
  int e = (int)(t % E);
  int b = (int)(t / E);
  const float* xs = x + ((long)b * NOUT + src[e]) * 32;
  float*       ad = agg + ((long)b * NOUT + dst[e]) * 64;
#pragma unroll
  for (int f = 0; f < 32; ++f) {
    float v = xs[f];
    atomicAdd(&ad[f], v);
    // IEEE float max via signed/unsigned atomic trick (init is -inf)
    if (v >= 0.0f) atomicMax((int*)&ad[32 + f], __float_as_int(v));
    else           atomicMin((unsigned int*)&ad[32 + f], __float_as_uint(v));
  }
}

// mean = has ? sum/cnt : x ; max = has ? max : x
__global__ void k_finalize(float* __restrict__ agg, const float* __restrict__ cnt,
                           const float* __restrict__ x, int NOUT, long total) {
  long t = blockIdx.x * (long)blockDim.x + threadIdx.x;
  if (t >= total) return;
  int  f  = (int)(t & 31);
  long bn = t >> 5;
  int  node = (int)(bn % NOUT);
  float c = cnt[node];
  float* a = agg + bn * 64;
  if (c > 0.0f) {
    a[f] = a[f] / c;
  } else {
    float xv = x[bn * 32 + f];
    a[f] = xv;
    a[32 + f] = xv;
  }
}

// ---------------------------------------------------------------------------
// WMMA MLP layer: x += W1 * relu(W0 * agg + b0) + b1  (per 16-node tile)
// Each wave owns one 16-row x 32-col tile. 6 v_wmma per tile.
// ---------------------------------------------------------------------------
__global__ void __launch_bounds__(256) k_mlp(
    float* __restrict__ x, const float* __restrict__ agg,
    const float* __restrict__ w0, const float* __restrict__ b0,
    const float* __restrict__ w1, const float* __restrict__ b1,
    int NOUT, int tiles)
{
  __shared__ __bf16 sw0[32 * 64];     // W0 converted to bf16
  __shared__ __bf16 sw1[32 * 32];     // W1 converted to bf16
  __shared__ __bf16 hst[8][16 * 32];  // per-wave relu(h) staging (layout swap)

  for (int i = threadIdx.x; i < 32 * 64; i += 256) sw0[i] = f2bf(w0[i]);
  for (int i = threadIdx.x; i < 32 * 32; i += 256) sw1[i] = f2bf(w1[i]);
  __syncthreads();

  const int wv   = threadIdx.x >> 5;
  const int lane = threadIdx.x & 31;
  const int tile = blockIdx.x * 8 + wv;
  if (tile >= tiles) return;                 // wave-uniform; EXEC stays all-1s

  const int ntPerB = NOUT >> 4;
  const int b    = tile / ntPerB;
  const int row0 = (tile % ntPerB) << 4;

  const int m    = lane & 15;       // A: row M ; B/C/D: col N
  const int half = lane >> 4;
  const int kbA  = half << 4;       // A K base within 32-chunk
  const int kbB  = half << 4;       // B K base within 32-chunk
  const int rb   = half << 3;       // C/D row base

  // ---- A = agg tile (16 x 64 f32 -> bf16), split into two K=32 chunks ----
  const float* ar = agg + (((long)b * NOUT + row0 + m) * 64);
  // prefetch next tile's row (L2-resident working set; hides L2 latency)
  __builtin_prefetch(ar + 16 * 64, 0, 1);
  v16bf a0, a1;
#pragma unroll
  for (int i = 0; i < 8; ++i) {
    a0[i]     = f2bf(ar[kbA + i]);
    a0[8 + i] = f2bf(ar[kbA + 8 + i]);
    a1[i]     = f2bf(ar[32 + kbA + i]);
    a1[8 + i] = f2bf(ar[32 + kbA + 8 + i]);
  }

  // ---- h = relu(A @ W0^T + b0), stored bf16 in LDS (16 x 32) ----
#pragma unroll
  for (int nt = 0; nt < 2; ++nt) {
    const __bf16* wp = &sw0[(nt * 16 + m) * 64 + kbB];   // B[k][n] = W0[n][k]
    v16bf bw0, bw1;
#pragma unroll
    for (int i = 0; i < 16; ++i) { bw0[i] = wp[i]; bw1[i] = wp[32 + i]; }
    v8f acc = {};
    acc = __builtin_amdgcn_wmma_f32_16x16x32_bf16(false, a0, false, bw0,
                                                  (short)0, acc, false, false);
    acc = __builtin_amdgcn_wmma_f32_16x16x32_bf16(false, a1, false, bw1,
                                                  (short)0, acc, false, false);
    float bias = b0[nt * 16 + m];
#pragma unroll
    for (int r = 0; r < 8; ++r) {
      float h = acc[r] + bias;
      h = h > 0.0f ? h : 0.0f;
      hst[wv][(rb + r) * 32 + nt * 16 + m] = f2bf(h);
    }
  }

  // wave-local LDS layout swap: C-layout -> A-layout
  __builtin_amdgcn_wave_barrier();
  asm volatile("s_wait_dscnt 0" ::: "memory");
  __builtin_amdgcn_wave_barrier();

  v16bf a2;
#pragma unroll
  for (int i = 0; i < 8; ++i) {
    a2[i]     = hst[wv][m * 32 + kbA + i];
    a2[8 + i] = hst[wv][m * 32 + kbA + 8 + i];
  }

  // ---- x += h @ W1^T + b1 ----
  float* xr = x + (((long)b * NOUT + row0) * 32);
#pragma unroll
  for (int nt = 0; nt < 2; ++nt) {
    const __bf16* wp = &sw1[(nt * 16 + m) * 32 + kbB];   // B[k][n] = W1[n][k]
    v16bf bw;
#pragma unroll
    for (int i = 0; i < 16; ++i) bw[i] = wp[i];
    v8f acc = {};
    acc = __builtin_amdgcn_wmma_f32_16x16x32_bf16(false, a2, false, bw,
                                                  (short)0, acc, false, false);
    float bias = b1[nt * 16 + m];
#pragma unroll
    for (int r = 0; r < 8; ++r) {
      int idx = (rb + r) * 32 + nt * 16 + m;
      xr[idx] += acc[r] + bias;
    }
  }
}

// ---------------------------------------------------------------------------
// Prologue: x0 = [relu(W_in @ [LN(x); goa] + b), embed]
// ---------------------------------------------------------------------------
__global__ void __launch_bounds__(128) k_input(
    const float* __restrict__ x, const float* __restrict__ goa,
    const float* __restrict__ gamma, const float* __restrict__ beta,
    const float* __restrict__ iw, const float* __restrict__ ib,
    const float* __restrict__ embed, float* __restrict__ x0,
    int NOUT, long total)
{
  __shared__ float sw[16 * 104];
  __shared__ float sg[40], sb[40], sib[16];
  for (int i = threadIdx.x; i < 16 * 104; i += 128) sw[i] = iw[i];
  if (threadIdx.x < 40) { sg[threadIdx.x] = gamma[threadIdx.x]; sb[threadIdx.x] = beta[threadIdx.x]; }
  if (threadIdx.x < 16) sib[threadIdx.x] = ib[threadIdx.x];
  __syncthreads();

  long t = blockIdx.x * (long)128 + threadIdx.x;
  if (t >= total) return;
  int node = (int)(t % NOUT);

  const float* xr = x + t * 40;
  float mean = 0.0f;
#pragma unroll
  for (int c = 0; c < 40; ++c) mean += xr[c];
  mean *= (1.0f / 40.0f);
  float var = 0.0f;
#pragma unroll
  for (int c = 0; c < 40; ++c) { float d = xr[c] - mean; var += d * d; }
  var *= (1.0f / 40.0f);
  float inv = rsqrtf(var + 1e-5f);

  float h[16];
#pragma unroll
  for (int j = 0; j < 16; ++j) h[j] = sib[j];
#pragma unroll 4
  for (int c = 0; c < 40; ++c) {
    float v = (xr[c] - mean) * inv * sg[c] + sb[c];
#pragma unroll
    for (int j = 0; j < 16; ++j) h[j] += sw[j * 104 + c] * v;
  }
  const float* gr = goa + t * 64;
#pragma unroll 4
  for (int c = 0; c < 64; ++c) {
    float v = gr[c];
#pragma unroll
    for (int j = 0; j < 16; ++j) h[j] += sw[j * 104 + 40 + c] * v;
  }

  float* o = x0 + t * 32;
  const float* em = embed + (long)node * 16;
#pragma unroll
  for (int j = 0; j < 16; ++j) o[j] = h[j] > 0.0f ? h[j] : 0.0f;
#pragma unroll
  for (int j = 0; j < 16; ++j) o[16 + j] = em[j];
}

// Epilogue: out (+)= dot(xd[node], clf_w[wofs:wofs+32]) (+ clf_b on init)
__global__ void k_dot(const float* __restrict__ xd, const float* __restrict__ clfw,
                      int wofs, const float* __restrict__ clfb,
                      float* __restrict__ out, long total, int initMode)
{
  long t = blockIdx.x * (long)blockDim.x + threadIdx.x;
  if (t >= total) return;
  const float* xr = xd + t * 32;
  float s = 0.0f;
#pragma unroll
  for (int f = 0; f < 32; ++f) s += xr[f] * clfw[wofs + f];
  if (initMode) out[t] = s + clfb[0];
  else          out[t] += s;
}

// ---------------------------------------------------------------------------
extern "C" void kernel_launch(void* const* d_in, const int* in_sizes, int n_in,
                              void* d_out, int out_size, void* d_ws, size_t ws_size,
                              hipStream_t stream) {
  (void)n_in; (void)out_size; (void)ws_size;
  const float* x     = (const float*)d_in[0];
  const float* goa   = (const float*)d_in[1];
  const int* dsts[3] = {(const int*)d_in[2], (const int*)d_in[4], (const int*)d_in[6]};
  const int* srcs[3] = {(const int*)d_in[3], (const int*)d_in[5], (const int*)d_in[7]};
  int        Es[3]   = {in_sizes[2], in_sizes[4], in_sizes[6]};
  const float* gamma = (const float*)d_in[8];
  const float* beta  = (const float*)d_in[9];
  const float* iw    = (const float*)d_in[10];
  const float* ib    = (const float*)d_in[11];
  const float* embed = (const float*)d_in[12];
  const float* w0s[3] = {(const float*)d_in[13], (const float*)d_in[17], (const float*)d_in[21]};
  const float* b0s[3] = {(const float*)d_in[14], (const float*)d_in[18], (const float*)d_in[22]};
  const float* w1s[3] = {(const float*)d_in[15], (const float*)d_in[19], (const float*)d_in[23]};
  const float* b1s[3] = {(const float*)d_in[16], (const float*)d_in[20], (const float*)d_in[24]};
  const float* clfw  = (const float*)d_in[25];
  const float* clfb  = (const float*)d_in[26];

  const int NOUT = in_sizes[12] / 16;          // embed is (NOUT, 16)
  const int B    = (int)((long)in_sizes[0] / ((long)NOUT * 40));
  const int L    = in_sizes[13] / (32 * 64);   // w0 is (L, 32, 64)
  const long BN  = (long)B * NOUT;

  // workspace: x0 | xcur | agg(sum+max) | cnt   (~123 MB, L2-resident)
  float* x0  = (float*)d_ws;
  float* xc  = x0 + BN * 32;
  float* agg = xc + BN * 32;
  float* cnt = agg + BN * 64;

  auto blocks = [](long n, int b) { return (unsigned int)((n + b - 1) / b); };

  k_input<<<blocks(BN, 128), 128, 0, stream>>>(x, goa, gamma, beta, iw, ib, embed,
                                               x0, NOUT, BN);
  k_dot<<<blocks(BN, 256), 256, 0, stream>>>(x0, clfw, 0, clfb, (float*)d_out, BN, 1);

  const int tiles = (int)(BN / 16);
  for (int d = 0; d < 3; ++d) {
    k_fill<<<blocks(NOUT, 256), 256, 0, stream>>>(cnt, 0.0f, NOUT);
    k_count<<<blocks(Es[d], 256), 256, 0, stream>>>(dsts[d], Es[d], cnt);
    hipMemcpyAsync(xc, x0, (size_t)BN * 32 * sizeof(float),
                   hipMemcpyDeviceToDevice, stream);
    for (int l = 0; l < L; ++l) {
      k_agg_init<<<blocks(BN * 64, 256), 256, 0, stream>>>(agg, BN * 64);
      k_scatter<<<blocks((long)Es[d] * B, 256), 256, 0, stream>>>(
          xc, dsts[d], srcs[d], Es[d], NOUT, (long)Es[d] * B, agg);
      k_finalize<<<blocks(BN * 32, 256), 256, 0, stream>>>(agg, cnt, xc, NOUT, BN * 32);
      k_mlp<<<blocks(tiles, 8), 256, 0, stream>>>(
          xc, agg,
          w0s[d] + (long)l * 32 * 64, b0s[d] + (long)l * 32,
          w1s[d] + (long)l * 32 * 32, b1s[d] + (long)l * 32,
          NOUT, tiles);
    }
    k_dot<<<blocks(BN, 256), 256, 0, stream>>>(xc, clfw, 32 * (d + 1), clfb,
                                               (float*)d_out, BN, 0);
  }
}
